// Mamba_10892037062806
// MI455X (gfx1250) — matrix operand order repaired and checked by reading, not compiled, not run on hardware
//
#include <hip/hip_runtime.h>
#include <hip/hip_bf16.h>
#include <math.h>

// ---------------- problem constants ----------------
#define NBLK   4
#define DDIM   128
#define NST    16
#define BATCH  4
#define SEQ    4096
#define ROWS   (BATCH*SEQ)     // 16384 tokens
#define INDIM  32
#define OUTDIM 32
#define CHUNKS 64
#define CLEN   (SEQ/CHUNKS)    // 64

typedef __attribute__((ext_vector_type(16))) __bf16 v16bf;
typedef __attribute__((ext_vector_type(8)))  float  v8f;

// ---------------- bf16 bit helpers (storage = ushort) ----------------
static __device__ inline unsigned short f2bf(float f) {
  unsigned int u = __float_as_uint(f);
  unsigned int r = u + 0x7FFFu + ((u >> 16) & 1u);   // round-to-nearest-even
  return (unsigned short)(r >> 16);
}
static __device__ inline __bf16 us2bf(unsigned short s) {
  union { unsigned short u; __bf16 b; } cv; cv.u = s; return cv.b;
}

// ---------------- fast transcendentals (hardware TRANS ops) ----------------
static __device__ inline float fast_rcp(float x) { return __builtin_amdgcn_rcpf(x); }
static __device__ inline float fast_softplus(float x) {
  float l = __logf(1.0f + __expf(x));          // v_exp_f32 + v_log_f32
  return (x > 20.f) ? x : l;
}
static __device__ inline float fast_sigmoid(float x) {
  return fast_rcp(1.0f + __expf(-x));
}
static __device__ inline float fast_tanh(float x) {
  float cx = fminf(15.f, fmaxf(-15.f, x));
  float e2 = __expf(2.0f * cx);
  return (e2 - 1.0f) * fast_rcp(e2 + 1.0f);
}

// ---------------- WMMA operand builders ----------------
// A: 16x32 bf16 tile from LDS buffer laid out [16][DDIM] (K-major per row).
// ISA layout: lanes 0-15 (m=lane): elems 0..7 -> K 0..7, elems 8..15 -> K 16..23
//             lanes 16-31 (m=lane-16): elems 0..7 -> K 8..15, elems 8..15 -> K 24..31
static __device__ inline v16bf load_a_tile(const unsigned short* zb, int lane, int kt) {
  const int half = lane >> 4;
  const int m    = lane & 15;
  const int kb0  = kt * 32 + half * 8;
  const int kb1  = kt * 32 + 16 + half * 8;
  v16bf a;
#pragma unroll
  for (int j = 0; j < 8; ++j) {
    a[j]     = us2bf(zb[m * DDIM + kb0 + j]);
    a[8 + j] = us2bf(zb[m * DDIM + kb1 + j]);
  }
  return a;
}
// B: 32x16 bf16 tile. Packed layout puts each lane's 16-element fragment
// contiguous (32 B, 32 B-aligned): dst[((nt*(K/32)+kt)*2+half)*16+n][16 j].
// One aligned vector load per fragment; wave reads 1 KB contiguous.
static __device__ inline v16bf load_b_tile(const unsigned short* pk, int lane,
                                           int nt, int kt, int K) {
  const int half = lane >> 4;
  const int n    = lane & 15;
  const size_t frag = ((size_t)(nt * (K / 32) + kt) * 2 + half) * 16 + n;
  return *(const v16bf*)(pk + frag * 16);
}

// ------ weight pack: src[K][Nc] f32 -> dst fragments (WMMA B layout, bf16) ------
// dst linear idx = ((((nt*(K/32) + kt)*2 + half)*16 + n)*16 + j)
// maps to src[k*Nc + col] with k = kt*32 + (j<8 ? half*8 + j : 16 + half*8 + (j-8)), col = nt*16 + n.
__global__ void pack_kernel(const float* __restrict__ src, unsigned short* __restrict__ dst,
                            int K, int Nc) {
  int idx = blockIdx.x * blockDim.x + threadIdx.x;
  if (idx >= K * Nc) return;
  int j    = idx & 15;
  int n    = (idx >> 4) & 15;
  int half = (idx >> 8) & 1;
  int kt   = (idx >> 9) % (K / 32);
  int nt   = (idx >> 9) / (K / 32);
  int k    = kt * 32 + (j < 8 ? half * 8 + j : 16 + half * 8 + (j - 8));
  int col  = nt * 16 + n;
  dst[idx] = f2bf(src[k * Nc + col]);
}

__global__ void aprep_kernel(const float* __restrict__ alog, float* __restrict__ A, int total) {
  int idx = blockIdx.x * blockDim.x + threadIdx.x;
  if (idx < total) A[idx] = -__expf(alog[idx]);
}

// ---------------- encoder: h = x @ enc_w + enc_b ----------------
__global__ __launch_bounds__(32) void enc_kernel(const float* __restrict__ x,
                                                 const float* __restrict__ w,
                                                 const float* __restrict__ b,
                                                 float* __restrict__ h) {
  __shared__ float xr[INDIM];
  const int row = blockIdx.x, lane = threadIdx.x;
  xr[lane] = x[row * INDIM + lane];
  __syncthreads();
#pragma unroll
  for (int dd = 0; dd < 4; ++dd) {
    int d = lane + dd * 32;
    float acc = b[d];
#pragma unroll
    for (int k = 0; k < INDIM; ++k) acc = fmaf(xr[k], w[k * DDIM + d], acc);
    h[row * DDIM + d] = acc;
  }
}

// ---------------- LN + delta/B/C projections (fused, WMMA) ----------------
__global__ __launch_bounds__(256) void ln_proj_kernel(
    const float* __restrict__ h,
    const float* __restrict__ nw, const float* __restrict__ nb,
    const unsigned short* __restrict__ Wdtp, const float* __restrict__ bdt,
    const unsigned short* __restrict__ WBp, const unsigned short* __restrict__ WCp,
    float* __restrict__ u, float* __restrict__ delta,
    float* __restrict__ Bm, float* __restrict__ Cm) {
  __shared__ unsigned short zb[16 * DDIM];
  __shared__ float red1[16][16];
  __shared__ float red2[16][16];
  __shared__ float stats[16][2];
  const int tid = threadIdx.x;
  const int rb  = blockIdx.x * 16;
  const int t   = tid >> 4, s = tid & 15;
  {  // get this wave's weight fragments in flight during the LN phase
    const int wv = tid >> 5, ln = tid & 31;
    __builtin_prefetch(Wdtp + ((size_t)(wv * 4) * 2 * 16 + ln) * 16, 0, 3);
  }
  float vals[8];
  float sum = 0.f, sq = 0.f;
#pragma unroll
  for (int j = 0; j < 8; ++j) {
    int d = s + j * 16;
    float v = h[(rb + t) * DDIM + d];
    vals[j] = v; sum += v; sq += v * v;
  }
  red1[t][s] = sum; red2[t][s] = sq;
  __syncthreads();
  if (s == 0) {
    float S = 0.f, Q = 0.f;
#pragma unroll
    for (int k = 0; k < 16; ++k) { S += red1[t][k]; Q += red2[t][k]; }
    float mu  = S * (1.0f / DDIM);
    float var = Q * (1.0f / DDIM) - mu * mu;
    stats[t][0] = mu; stats[t][1] = rsqrtf(var + 1e-5f);
  }
  __syncthreads();
  const float mu = stats[t][0], rs = stats[t][1];
#pragma unroll
  for (int j = 0; j < 8; ++j) {
    int d = s + j * 16;
    float z = (vals[j] - mu) * rs * nw[d] + nb[d];
    zb[t * DDIM + d] = f2bf(z);
    u[(rb + t) * DDIM + d] = z;
  }
  __syncthreads();
  const int wave = tid >> 5, lane = tid & 31;
  // delta pre-activation: wave w owns output cols [16w, 16w+16)
  v8f acc = {};
#pragma unroll
  for (int kt = 0; kt < 4; ++kt) {
    v16bf a = load_a_tile(zb, lane, kt);
    v16bf b = load_b_tile(Wdtp, lane, wave, kt, DDIM);
    acc = __builtin_amdgcn_wmma_f32_16x16x32_bf16(false, a, false, b, (short)0, acc, false, false);
  }
  const int half = lane >> 4, n = lane & 15;
#pragma unroll
  for (int r = 0; r < 8; ++r) {
    int m   = r + half * 8;
    int col = wave * 16 + n;
    delta[(rb + m) * DDIM + col] = fast_softplus(acc[r] + bdt[col]);
  }
  // waves 0/1 additionally compute Bm / Cm (N = 16, one tile each)
  if (wave < 2) {
    const unsigned short* pk = (wave == 0) ? WBp : WCp;
    v8f a2 = {};
#pragma unroll
    for (int kt = 0; kt < 4; ++kt) {
      v16bf a = load_a_tile(zb, lane, kt);
      v16bf b = load_b_tile(pk, lane, 0, kt, DDIM);
      a2 = __builtin_amdgcn_wmma_f32_16x16x32_bf16(false, a, false, b, (short)0, a2, false, false);
    }
    float* dst = (wave == 0) ? Bm : Cm;
#pragma unroll
    for (int r = 0; r < 8; ++r) {
      int m = r + half * 8;
      dst[(rb + m) * NST + n] = a2[r];
    }
  }
}

// ---------------- chunked associative scan (3 phases) ----------------
// lane mapping per 32-thread wave: d = 2*dp + (lane>=16), n = lane&15
__global__ __launch_bounds__(32) void scan1_kernel(
    const float* __restrict__ delta, const float* __restrict__ u,
    const float* __restrict__ Bm, const float* __restrict__ A,
    float* __restrict__ P, float* __restrict__ S) {
  const int bid = blockIdx.x;
  const int ch  = bid % CHUNKS;
  const int dp  = (bid / CHUNKS) % (DDIM / 2);
  const int b   = bid / (CHUNKS * (DDIM / 2));
  const int lane = threadIdx.x;
  const int d = dp * 2 + (lane >> 4);
  const int n = lane & 15;
  const float Av = A[d * NST + n];
  float Pv = 1.f, Sv = 0.f;
  const int l0 = ch * CLEN;
  for (int l = l0; l < l0 + CLEN; ++l) {
    int base = b * SEQ + l;
    float dl = delta[base * DDIM + d];
    float ul = u[base * DDIM + d];
    float bm = Bm[base * NST + n];
    float dA = __expf(dl * Av);                 // v_exp_f32 on the critical chain
    Sv = dA * Sv + dl * bm * ul;
    Pv *= dA;
  }
  const int o = ((b * DDIM + d) * CHUNKS + ch) * NST + n;
  P[o] = Pv; S[o] = Sv;
}

__global__ __launch_bounds__(32) void scan2_kernel(
    const float* __restrict__ P, const float* __restrict__ S, float* __restrict__ H) {
  const int bid = blockIdx.x;
  const int dp  = bid % (DDIM / 2);
  const int b   = bid / (DDIM / 2);
  const int lane = threadIdx.x;
  const int d = dp * 2 + (lane >> 4);
  const int n = lane & 15;
  float hin = 0.f;
  for (int ch = 0; ch < CHUNKS; ++ch) {
    int o = ((b * DDIM + d) * CHUNKS + ch) * NST + n;
    H[o] = hin;
    hin = P[o] * hin + S[o];
  }
}

__global__ __launch_bounds__(32) void scan3_kernel(
    const float* __restrict__ delta, const float* __restrict__ u,
    const float* __restrict__ Bm, const float* __restrict__ Cm,
    const float* __restrict__ A, const float* __restrict__ H,
    const float* __restrict__ Dpv, float* __restrict__ y) {
  const int bid = blockIdx.x;
  const int ch  = bid % CHUNKS;
  const int dp  = (bid / CHUNKS) % (DDIM / 2);
  const int b   = bid / (CHUNKS * (DDIM / 2));
  const int lane = threadIdx.x;
  const int d = dp * 2 + (lane >> 4);
  const int n = lane & 15;
  const float Av = A[d * NST + n];
  const float Dv = Dpv[d];
  float hst = H[((b * DDIM + d) * CHUNKS + ch) * NST + n];
  const int l0 = ch * CLEN;
  for (int l = l0; l < l0 + CLEN; ++l) {
    int base = b * SEQ + l;
    float dl = delta[base * DDIM + d];
    float ul = u[base * DDIM + d];
    float bm = Bm[base * NST + n];
    float cm = Cm[base * NST + n];
    float dA = __expf(dl * Av);
    hst = dA * hst + dl * bm * ul;
    float c = hst * cm;
    c += __shfl_down(c, 8, 16);
    c += __shfl_down(c, 4, 16);
    c += __shfl_down(c, 2, 16);
    c += __shfl_down(c, 1, 16);
    if (n == 0) y[base * DDIM + d] = c + Dv * ul;
  }
}

// ---------------- GELU + GLU GEMM + residual (fused, WMMA) ----------------
__global__ __launch_bounds__(256) void glu_kernel(
    const float* __restrict__ y, float* __restrict__ h,
    const unsigned short* __restrict__ glup, const float* __restrict__ glub) {
  __shared__ unsigned short zb[16 * DDIM];
  const int tid = threadIdx.x;
  const int rb  = blockIdx.x * 16;
  const int t = tid >> 4, s = tid & 15;
  {
    const int wv = tid >> 5, ln = tid & 31;
    __builtin_prefetch(glup + ((size_t)(wv * 4) * 2 * 16 + ln) * 16, 0, 3);
    __builtin_prefetch(glup + ((size_t)((wv + 8) * 4) * 2 * 16 + ln) * 16, 0, 3);
  }
#pragma unroll
  for (int j = 0; j < 8; ++j) {
    int d = s + j * 16;
    float v = y[(rb + t) * DDIM + d];
    float g = 0.5f * v * (1.0f + erff(v * 0.70710678118654752f));  // exact GELU
    zb[t * DDIM + d] = f2bf(g);
  }
  __syncthreads();
  const int wave = tid >> 5, lane = tid & 31;
  v8f accA = {}, accG = {};
#pragma unroll
  for (int kt = 0; kt < 4; ++kt) {
    v16bf a  = load_a_tile(zb, lane, kt);
    v16bf bA = load_b_tile(glup, lane, wave, kt, DDIM);        // value half
    v16bf bG = load_b_tile(glup, lane, wave + 8, kt, DDIM);    // gate half
    accA = __builtin_amdgcn_wmma_f32_16x16x32_bf16(false, a, false, bA, (short)0, accA, false, false);
    accG = __builtin_amdgcn_wmma_f32_16x16x32_bf16(false, a, false, bG, (short)0, accG, false, false);
  }
  const int half = lane >> 4, n = lane & 15;
#pragma unroll
  for (int r = 0; r < 8; ++r) {
    int m   = r + half * 8;
    int col = wave * 16 + n;
    float av  = accA[r] + glub[col];
    float gv  = accG[r] + glub[DDIM + col];
    float val = av * fast_sigmoid(gv);
    int idx = (rb + m) * DDIM + col;
    h[idx] = val + h[idx];                                     // residual
  }
}

// ---------------- decoder GEMM + tanh (WMMA) ----------------
__global__ __launch_bounds__(64) void dec_kernel(
    const float* __restrict__ h, const unsigned short* __restrict__ decp,
    const float* __restrict__ decb, float* __restrict__ out) {
  __shared__ unsigned short hb[16 * DDIM];
  const int tid = threadIdx.x;
  const int rb  = blockIdx.x * 16;
  {
    const int wv = tid >> 5, ln = tid & 31;
    __builtin_prefetch(decp + ((size_t)(wv * 4) * 2 * 16 + ln) * 16, 0, 3);
  }
  for (int idx = tid; idx < 16 * DDIM; idx += 64)
    hb[idx] = f2bf(h[rb * DDIM + idx]);
  __syncthreads();
  const int wave = tid >> 5, lane = tid & 31;
  v8f acc = {};
#pragma unroll
  for (int kt = 0; kt < 4; ++kt) {
    v16bf a = load_a_tile(hb, lane, kt);
    v16bf b = load_b_tile(decp, lane, wave, kt, DDIM);
    acc = __builtin_amdgcn_wmma_f32_16x16x32_bf16(false, a, false, b, (short)0, acc, false, false);
  }
  const int half = lane >> 4, n = lane & 15;
#pragma unroll
  for (int r = 0; r < 8; ++r) {
    int m   = r + half * 8;
    int col = wave * 16 + n;
    out[(rb + m) * OUTDIM + col] = fast_tanh(acc[r] + decb[col]);
  }
}

// ---------------- host driver ----------------
extern "C" void kernel_launch(void* const* d_in, const int* in_sizes, int n_in,
                              void* d_out, int out_size, void* d_ws, size_t ws_size,
                              hipStream_t stream) {
  const float* x      = (const float*)d_in[0];
  const float* enc_w  = (const float*)d_in[1];
  const float* enc_b  = (const float*)d_in[2];
  const float* norm_w = (const float*)d_in[3];
  const float* norm_b = (const float*)d_in[4];
  const float* A_log  = (const float*)d_in[5];
  const float* Dp     = (const float*)d_in[6];
  const float* Wdt    = (const float*)d_in[7];
  const float* bdt    = (const float*)d_in[8];
  const float* WB     = (const float*)d_in[9];
  const float* WC     = (const float*)d_in[10];
  const float* glu_w  = (const float*)d_in[11];
  const float* glu_b  = (const float*)d_in[12];
  const float* dec_w  = (const float*)d_in[13];
  const float* dec_b  = (const float*)d_in[14];

  char* ws = (char*)d_ws;
  size_t off = 0;
  auto alloc = [&](size_t bytes) -> char* {
    off = (off + 255) & ~(size_t)255;
    char* p = ws + off;
    off += bytes;
    return p;
  };
  float* h     = (float*)alloc((size_t)ROWS * DDIM * 4);
  float* u     = (float*)alloc((size_t)ROWS * DDIM * 4);
  float* delta = (float*)alloc((size_t)ROWS * DDIM * 4);
  float* ybuf  = (float*)alloc((size_t)ROWS * DDIM * 4);
  float* Bm    = (float*)alloc((size_t)ROWS * NST * 4);
  float* Cm    = (float*)alloc((size_t)ROWS * NST * 4);
  float* P     = (float*)alloc((size_t)BATCH * DDIM * CHUNKS * NST * 4);
  float* S     = (float*)alloc((size_t)BATCH * DDIM * CHUNKS * NST * 4);
  float* H     = (float*)alloc((size_t)BATCH * DDIM * CHUNKS * NST * 4);
  float* Afl   = (float*)alloc((size_t)NBLK * DDIM * NST * 4);
  unsigned short* Wdtp = (unsigned short*)alloc((size_t)NBLK * DDIM * DDIM * 2);
  unsigned short* glup = (unsigned short*)alloc((size_t)NBLK * DDIM * 2 * DDIM * 2);
  unsigned short* WBp  = (unsigned short*)alloc((size_t)NBLK * DDIM * NST * 2);
  unsigned short* WCp  = (unsigned short*)alloc((size_t)NBLK * DDIM * NST * 2);
  unsigned short* decp = (unsigned short*)alloc((size_t)DDIM * OUTDIM * 2);

  // --- weight packing (bf16, WMMA B fragment layout) + A = -exp(A_log) ---
  for (int i = 0; i < NBLK; ++i) {
    pack_kernel<<<(DDIM * DDIM + 255) / 256, 256, 0, stream>>>(
        Wdt + (size_t)i * DDIM * DDIM, Wdtp + (size_t)i * DDIM * DDIM, DDIM, DDIM);
    pack_kernel<<<(DDIM * 2 * DDIM + 255) / 256, 256, 0, stream>>>(
        glu_w + (size_t)i * DDIM * 2 * DDIM, glup + (size_t)i * DDIM * 2 * DDIM, DDIM, 2 * DDIM);
    pack_kernel<<<(DDIM * NST + 255) / 256, 256, 0, stream>>>(
        WB + (size_t)i * DDIM * NST, WBp + (size_t)i * DDIM * NST, DDIM, NST);
    pack_kernel<<<(DDIM * NST + 255) / 256, 256, 0, stream>>>(
        WC + (size_t)i * DDIM * NST, WCp + (size_t)i * DDIM * NST, DDIM, NST);
  }
  pack_kernel<<<(DDIM * OUTDIM + 255) / 256, 256, 0, stream>>>(dec_w, decp, DDIM, OUTDIM);
  aprep_kernel<<<(NBLK * DDIM * NST + 255) / 256, 256, 0, stream>>>(A_log, Afl, NBLK * DDIM * NST);

  // --- encoder ---
  enc_kernel<<<ROWS, 32, 0, stream>>>(x, enc_w, enc_b, h);

  // --- residual blocks ---
  for (int i = 0; i < NBLK; ++i) {
    ln_proj_kernel<<<ROWS / 16, 256, 0, stream>>>(
        h, norm_w + i * DDIM, norm_b + i * DDIM,
        Wdtp + (size_t)i * DDIM * DDIM, bdt + i * DDIM,
        WBp + (size_t)i * DDIM * NST, WCp + (size_t)i * DDIM * NST,
        u, delta, Bm, Cm);
    scan1_kernel<<<BATCH * (DDIM / 2) * CHUNKS, 32, 0, stream>>>(
        delta, u, Bm, Afl + i * DDIM * NST, P, S);
    scan2_kernel<<<BATCH * (DDIM / 2), 32, 0, stream>>>(P, S, H);
    scan3_kernel<<<BATCH * (DDIM / 2) * CHUNKS, 32, 0, stream>>>(
        delta, u, Bm, Cm, Afl + i * DDIM * NST, H, Dp + i * DDIM, ybuf);
    glu_kernel<<<ROWS / 16, 256, 0, stream>>>(
        ybuf, h, glup + (size_t)i * DDIM * 2 * DDIM, glu_b + i * 2 * DDIM);
  }

  // --- decoder ---
  dec_kernel<<<ROWS / 16, 64, 0, stream>>>(h, decp, dec_b, (float*)d_out);
}
